// KeyValueAttention_74268574482972
// MI455X (gfx1250) — matrix operand order repaired and checked by reading, not compiled, not run on hardware
//
#include <hip/hip_runtime.h>
#include <hip/hip_bf16.h>

typedef __attribute__((ext_vector_type(16))) _Float16 v16h;
typedef __attribute__((ext_vector_type(8)))  _Float16 v8h;
typedef __attribute__((ext_vector_type(8)))  float    v8f;

constexpr int Bn  = 8;
constexpr int QL  = 1024;
constexpr int SL  = 2048;
constexpr int QIN = 1024;
constexpr int QK  = 512;
constexpr int VD  = 512;
constexpr int ON  = 1024;

constexpr int KSTEP = 64;               // K chunk staged per buffer
constexpr int CSTR  = 72;               // halfs per B-tile column in LDS (64 data + 8 pad)
constexpr int BUFH  = 64 * CSTR;        // halfs per buffer (4608) -> 9216 B, x2 buffers

__device__ __forceinline__ v8f wmma_f16(v16h a, v16h b, v8f c) {
  // D = A(16x32 f16) * B(32x16 f16) + C(16x16 f32)
  return __builtin_amdgcn_wmma_f32_16x16x32_f16(false, a, false, b, (short)0, c, false, false);
}

// A-fragment (16x32 f16): lanes 0-15 hold K[0..7]+K[16..23] of row M=lane,
// lanes 16-31 hold K[8..15]+K[24..31].  `p` is pre-offset by koffA.
__device__ __forceinline__ v16h load_afrag(const _Float16* p) {
  v8h lo = *reinterpret_cast<const v8h*>(p);
  v8h hi = *reinterpret_cast<const v8h*>(p + 16);
  v16h r;
#pragma unroll
  for (int i = 0; i < 8; ++i) { r[i] = lo[i]; r[i + 8] = hi[i]; }
  return r;
}

// Same fragment but converting fp32 -> f16 in registers.
__device__ __forceinline__ v16h load_afrag(const float* p) {
  v16h r;
#pragma unroll
  for (int i = 0; i < 8; ++i) { r[i] = (_Float16)p[i]; r[i + 8] = (_Float16)p[16 + i]; }
  return r;
}

// 16 contiguous halfs, only 16-byte aligned (unused).
__device__ __forceinline__ v16h load_contig16(const _Float16* p) {
  v8h lo = *reinterpret_cast<const v8h*>(p);
  v8h hi = *reinterpret_cast<const v8h*>(p + 8);
  v16h r;
#pragma unroll
  for (int i = 0; i < 8; ++i) { r[i] = lo[i]; r[i + 8] = hi[i]; }
  return r;
}

// Async global -> LDS copy of 16 bytes per lane (one instruction per wave, ASYNCcnt).
__device__ __forceinline__ void async_copy_b128(unsigned lds_byte_addr, const _Float16* gsrc) {
  asm volatile("global_load_async_to_lds_b128 %0, %1, off"
               :
               : "v"(lds_byte_addr), "v"(gsrc)
               : "memory");
}
__device__ __forceinline__ void wait_async0() {
  asm volatile("s_wait_asynccnt 0x0" ::: "memory");
}

// Wt[n][k] = (f16) W[k][n]
__global__ void convert_transpose(const float* __restrict__ W, _Float16* __restrict__ Wt,
                                  int K, int N) {
  int k = blockIdx.x * 16 + threadIdx.x;
  int n = blockIdx.y * 16 + threadIdx.y;
  if (k < K && n < N) Wt[(size_t)n * K + k] = (_Float16)W[(size_t)k * N + n];
}

// C[M][N] = A[M][K] @ Bt[N][K]^T + bias[N]   (Bt is B stored transposed, contiguous along K)
// Block: 256 threads = 8 waves; wave computes 32x64 (8 accumulators); block covers 256(M) x 64(N).
// B tile (64 cols x 64 K) shared by all 8 waves via double-buffered LDS, filled with
// async global->LDS copies; stage(k+1) overlaps compute(k); one barrier per 16 WMMAs.
// Grid: x = N/64, y = M/256, z = batch.
template <typename AT, typename CT, bool STORE_T>
__global__ __launch_bounds__(256) void gemm_wmma(
    const AT* __restrict__ A, const _Float16* __restrict__ Bt,
    const float* __restrict__ bias, CT* __restrict__ C,
    int K, int lda, int ldb, int ldc,
    long long sA, long long sB, long long sC) {
  __shared__ _Float16 Bs[2][BUFH];

  const int wave = threadIdx.x >> 5;
  const int lane = threadIdx.x & 31;
  const AT*       Ab = A  + (long long)blockIdx.z * sA;
  const _Float16* Bb = Bt + (long long)blockIdx.z * sB;
  CT*             Cb = C  + (long long)blockIdx.z * sC;

  const int row0 = blockIdx.y * 256 + wave * 32;
  const int n0b  = blockIdx.x * 64;
  const int m    = lane & 15;
  const int hi   = (lane >> 4) & 1;
  const int koffA = hi ? 8 : 0;    // A layout split
  const int koffB = hi ? 8 : 0;    // B layout split
  const int mb    = hi ? 8 : 0;    // C/D row split

  // Staging assignment: thread t copies 16 halfs of column sc at k-offset sk (2x b128).
  const int t  = threadIdx.x;
  const int sc = t >> 2;           // 0..63
  const int sk = (t & 3) * 16;     // 0,16,32,48
  const _Float16* gsrc = Bb + (size_t)(n0b + sc) * ldb + sk;
  const unsigned lds0 =
      (unsigned)(unsigned long long)&Bs[0][0] + (unsigned)((sc * CSTR + sk) * 2);

  v8f acc[2][4] = {};
  const AT* arow0 = Ab + (size_t)(row0 + m) * lda + koffA;
  const AT* arow1 = arow0 + (size_t)16 * lda;
  const _Float16* bfragbase = &Bs[0][0] + (size_t)m * CSTR + koffB;

  // prologue: stage k0 = 0 into buffer 0
  async_copy_b128(lds0, gsrc);
  async_copy_b128(lds0 + 16, gsrc + 8);
  int buf = 0;
  for (int k0 = 0; k0 < K; k0 += KSTEP) {
    wait_async0();                                // this wave's stage into `buf` done
    __syncthreads();                              // all waves staged; prior reads done
    if (k0 + KSTEP < K) {
      const unsigned d = lds0 + (unsigned)((buf ^ 1) * BUFH * 2);
      const _Float16* g = gsrc + k0 + KSTEP;
      async_copy_b128(d, g);
      async_copy_b128(d + 16, g + 8);
    }
#pragma unroll
    for (int c2 = 0; c2 < 2; ++c2) {              // two 32-wide K chunks per buffer
      const int kk = k0 + c2 * 32;
      v16h a0 = load_afrag(arow0 + kk);
      v16h a1 = load_afrag(arow1 + kk);
      const _Float16* bb = bfragbase + (size_t)buf * BUFH + c2 * 32;
#pragma unroll
      for (int j = 0; j < 4; ++j) {
        v16h bj = load_afrag(bb + j * 16 * CSTR);
        acc[0][j] = wmma_f16(a0, bj, acc[0][j]);
        acc[1][j] = wmma_f16(a1, bj, acc[1][j]);
      }
    }
    buf ^= 1;
  }
#pragma unroll
  for (int mt = 0; mt < 2; ++mt) {
    const int rbase = row0 + mt * 16 + mb;
#pragma unroll
    for (int j = 0; j < 4; ++j) {
      const int col = n0b + j * 16 + m;
      const float bv = bias ? bias[col] : 0.0f;
#pragma unroll
      for (int r = 0; r < 8; ++r) {
        float val = acc[mt][j][r] + bv;
        if constexpr (STORE_T)
          Cb[(size_t)col * ldc + (rbase + r)] = (CT)val;   // C^T store (for V^T)
        else
          Cb[(size_t)(rbase + r) * ldc + col] = (CT)val;
      }
    }
  }
}

// score = (qp @ k^T)/sqrt(QK), mask, row-softmax over SL, write attn_w (f32).
// One block handles 16 query rows of one batch; full 16xSL f32 strip lives in LDS.
// Waves own disjoint n-tiles (no cross-wave B reuse), so B comes straight from global/L2.
__global__ __launch_bounds__(256) void score_softmax(
    const _Float16* __restrict__ qph, const _Float16* __restrict__ kh,
    const int* __restrict__ mask, float* __restrict__ attn_w) {
  extern __shared__ float smem[];
  float* S      = smem;              // 16 * SL
  float* redmax = smem + 16 * SL;    // 256
  float* redsum = redmax + 256;      // 256

  const int b    = blockIdx.y;
  const int q0   = blockIdx.x * 16;
  const int wave = threadIdx.x >> 5;
  const int lane = threadIdx.x & 31;
  const int m    = lane & 15;
  const int hi   = (lane >> 4) & 1;
  const int koffA = hi ? 8 : 0;
  const int koffB = hi ? 8 : 0;
  const int mb    = hi ? 8 : 0;

  const _Float16* arow = qph + ((size_t)b * QL + q0 + m) * QK + koffA;
  const _Float16* kb   = kh + (size_t)b * SL * QK;
  const float inv_scale = 0.04419417382415922f;  // 1/sqrt(512)

  for (int nt = wave; nt < SL / 16; nt += 8) {
    const int n0 = nt * 16;
    const _Float16* brow = kb + (size_t)(n0 + m) * QK + koffB;
    v8f acc = {};
    for (int k0 = 0; k0 < QK; k0 += 32) {
      v16h a  = load_afrag(arow + k0);
      v16h bf = load_afrag(brow + k0);
      acc = wmma_f16(a, bf, acc);
    }
    const int mv = mask[b * SL + n0 + m];
#pragma unroll
    for (int r = 0; r < 8; ++r) {
      float sv = acc[r] * inv_scale;
      sv = (mv == 0) ? -1.0e10f : sv;
      S[(mb + r) * SL + n0 + m] = sv;
    }
  }
  __syncthreads();

  // Softmax: 16 threads per row, 16 rows.
  const int r = threadIdx.x >> 4;
  const int t = threadIdx.x & 15;
  float mx = -3.0e38f;
  for (int c = t; c < SL; c += 16) mx = fmaxf(mx, S[r * SL + c]);
  redmax[r * 16 + t] = mx;
  __syncthreads();
  mx = redmax[r * 16];
#pragma unroll
  for (int i = 1; i < 16; ++i) mx = fmaxf(mx, redmax[r * 16 + i]);

  float sum = 0.0f;
  for (int c = t; c < SL; c += 16) {
    float e = __expf(S[r * SL + c] - mx);
    S[r * SL + c] = e;
    sum += e;
  }
  redsum[r * 16 + t] = sum;
  __syncthreads();
  float tot = 0.0f;
#pragma unroll
  for (int i = 0; i < 16; ++i) tot += redsum[r * 16 + i];
  const float inv = 1.0f / tot;

  const size_t rowbase = ((size_t)b * QL + q0 + r) * SL;
  for (int c = t; c < SL; c += 16) attn_w[rowbase + c] = S[r * SL + c] * inv;
}

extern "C" void kernel_launch(void* const* d_in, const int* in_sizes, int n_in,
                              void* d_out, int out_size, void* d_ws, size_t ws_size,
                              hipStream_t stream) {
  (void)in_sizes; (void)n_in; (void)out_size; (void)ws_size;
  const float* q    = (const float*)d_in[0];
  const float* s    = (const float*)d_in[1];
  const int*   mask = (const int*)d_in[2];
  const float* Wq   = (const float*)d_in[3];
  const float* bq   = (const float*)d_in[4];
  const float* Wk   = (const float*)d_in[5];
  const float* bk   = (const float*)d_in[6];
  const float* Wv   = (const float*)d_in[7];
  const float* bv   = (const float*)d_in[8];
  const float* Wo   = (const float*)d_in[9];
  const float* bo   = (const float*)d_in[10];

  float* attn_w   = (float*)d_out;                          // [B, QL, SL]
  float* attn_out = attn_w + (size_t)Bn * QL * SL;          // [B, QL, ON]

  char* ws = (char*)d_ws;
  size_t off = 0;
  auto alloc = [&](size_t bytes) {
    char* p = ws + off;
    off += (bytes + 255) & ~(size_t)255;
    return p;
  };
  _Float16* WqT = (_Float16*)alloc((size_t)QK * QIN * 2);      // [QK][QIN]
  _Float16* WkT = (_Float16*)alloc((size_t)QK * QIN * 2);      // [QK][QIN]
  _Float16* WvT = (_Float16*)alloc((size_t)VD * QIN * 2);      // [VD][QIN]
  _Float16* WoT = (_Float16*)alloc((size_t)ON * VD * 2);       // [ON][VD]
  _Float16* qph = (_Float16*)alloc((size_t)Bn * QL * QK * 2);  // [B*QL][QK]
  _Float16* khb = (_Float16*)alloc((size_t)Bn * SL * QK * 2);  // [B*SL][QK]
  _Float16* vhT = (_Float16*)alloc((size_t)Bn * VD * SL * 2);  // [B][VD][SL]
  _Float16* aoh = (_Float16*)alloc((size_t)Bn * QL * VD * 2);  // [B*QL][VD]

  dim3 tb(16, 16);
  convert_transpose<<<dim3(QIN / 16, QK / 16), tb, 0, stream>>>(Wq, WqT, QIN, QK);
  convert_transpose<<<dim3(QIN / 16, QK / 16), tb, 0, stream>>>(Wk, WkT, QIN, QK);
  convert_transpose<<<dim3(QIN / 16, VD / 16), tb, 0, stream>>>(Wv, WvT, QIN, VD);
  convert_transpose<<<dim3(VD / 16, ON / 16), tb, 0, stream>>>(Wo, WoT, VD, ON);

  // qp = q @ Wq + bq   (f32 A converted in-flight)
  gemm_wmma<float, _Float16, false>
      <<<dim3(QK / 64, (Bn * QL) / 256, 1), 256, 0, stream>>>(
          q, WqT, bq, qph, QIN, QIN, QIN, QK, 0, 0, 0);
  // k = s @ Wk + bk
  gemm_wmma<float, _Float16, false>
      <<<dim3(QK / 64, (Bn * SL) / 256, 1), 256, 0, stream>>>(
          s, WkT, bk, khb, QIN, QIN, QIN, QK, 0, 0, 0);
  // v^T = (s @ Wv + bv)^T per batch -> [B][VD][SL]
  gemm_wmma<float, _Float16, true>
      <<<dim3(VD / 64, SL / 256, Bn), 256, 0, stream>>>(
          s, WvT, bv, vhT, QIN, QIN, QIN, SL,
          (long long)SL * QIN, 0, (long long)VD * SL);
  // score + mask + softmax -> attn_w (f32 output)
  score_softmax<<<dim3(QL / 16, Bn), 256, (size_t)(16 * SL + 512) * sizeof(float), stream>>>(
      qph, khb, mask, attn_w);
  // attn_out_pre = attn_w @ v   (A = f32 attn_w from d_out, B = v^T)
  gemm_wmma<float, _Float16, false>
      <<<dim3(VD / 64, QL / 256, Bn), 256, 0, stream>>>(
          attn_w, vhT, nullptr, aoh, SL, SL, SL, VD,
          (long long)QL * SL, (long long)VD * SL, (long long)QL * VD);
  // attn_out = attn_out_pre @ Wo + bo  (f32 output)
  gemm_wmma<_Float16, float, false>
      <<<dim3(ON / 64, (Bn * QL) / 256, 1), 256, 0, stream>>>(
          aoh, WoT, bo, attn_out, VD, VD, VD, ON, 0, 0, 0);
}
